// MultiHeadSelfAttention_59176059404570
// MI455X (gfx1250) — matrix operand order repaired and checked by reading, not compiled, not run on hardware
//
#include <hip/hip_runtime.h>
#include <hip/hip_bf16.h>

// ---------------- problem constants (match reference) ----------------
constexpr int Bc  = 4;
constexpr int Sc  = 2048;
constexpr int Ec  = 1024;
constexpr int Hc  = 16;
constexpr int HDc = 64;   // head dim

// ---------------- vector types for WMMA ----------------
typedef __attribute__((ext_vector_type(16))) __bf16 v16bf;
typedef __attribute__((ext_vector_type(8)))  float  v8f;

// D = A(16x32 bf16) * B(32x16 bf16) + C(16x16 f32)
__device__ __forceinline__ v8f wmma_bf16(v16bf a, v16bf b, v8f c) {
  return __builtin_amdgcn_wmma_f32_16x16x32_bf16(
      false, a, false, b, (short)0, c, false, false);
}

// =====================================================================
// Kernel 1: per-head QKV projection.
//   Q,K stored bf16 [B,H,S,HD]; V stored transposed bf16 [B,H,HD,S].
//   Block = 128 thr (4 waves), each wave computes one 16x64 tile per matrix.
// =====================================================================
__global__ __launch_bounds__(128) void qkv_kernel(
    const float* __restrict__ x,
    const float* __restrict__ Wq, const float* __restrict__ Wk,
    const float* __restrict__ Wv,
    __bf16* __restrict__ Qb, __bf16* __restrict__ Kb,
    __bf16* __restrict__ Vt)
{
  __shared__ __bf16 WtLds[3][HDc * HDc];   // W^T in bf16: [n][k]
  const int bh = blockIdx.y;
  const int b  = bh / Hc;
  const int h  = bh % Hc;
  const int tid = threadIdx.x;

  const float* Ws[3] = { Wq + (size_t)h*HDc*HDc,
                         Wk + (size_t)h*HDc*HDc,
                         Wv + (size_t)h*HDc*HDc };
  for (int w = 0; w < 3; ++w)
    for (int i = tid; i < HDc*HDc; i += 128) {
      int k = i >> 6, n = i & 63;                  // i = k*64 + n
      WtLds[w][n*HDc + k] = (__bf16)Ws[w][i];      // native cvt f32->bf16
    }
  __syncthreads();

  const int lane = tid & 31;
  const int wave = tid >> 5;
  const int row0 = lane & 15;          // A-row / B-col / C-col within tile
  const int lh   = lane >> 4;          // lane half
  const int mrow = blockIdx.x*64 + wave*16;        // tile base (s index)
  const int srow = mrow + row0;

  // A tiles: x[b, srow, h*64 + 0..63] -> two 16x32 bf16 chunks (A layout)
  v16bf a[2];
  const float* xr = x + ((size_t)b * Sc + srow) * Ec + h * HDc;
  #pragma unroll
  for (int kc = 0; kc < 2; ++kc) {
    const int base = kc*32 + lh*8;
    #pragma unroll
    for (int e = 0; e < 8; ++e) {
      a[kc][e]     = (__bf16)xr[base + e];
      a[kc][e + 8] = (__bf16)xr[base + 16 + e];
    }
  }

  #pragma unroll
  for (int w = 0; w < 3; ++w) {
    const __bf16* Wt = &WtLds[w][0];
    v8f acc[4] = {};
    #pragma unroll
    for (int kc = 0; kc < 2; ++kc) {
      #pragma unroll
      for (int nt = 0; nt < 4; ++nt) {
        const __bf16* p = Wt + (nt*16 + row0)*HDc + kc*32 + lh*8;
        v16bf bt;
        #pragma unroll
        for (int e = 0; e < 8; ++e) { bt[e] = p[e]; bt[e+8] = p[16+e]; }
        acc[nt] = wmma_bf16(a[kc], bt, acc[nt]);
      }
    }
    if (w < 2) {
      __bf16* dst = (w == 0 ? Qb : Kb) + (size_t)bh * Sc * HDc;
      #pragma unroll
      for (int nt = 0; nt < 4; ++nt)
        #pragma unroll
        for (int v = 0; v < 8; ++v) {
          int m = mrow + v + 8*lh;
          dst[(size_t)m*HDc + nt*16 + row0] = (__bf16)acc[nt][v];
        }
    } else {
      __bf16* dst = Vt + (size_t)bh * HDc * Sc;      // [d][s]
      #pragma unroll
      for (int nt = 0; nt < 4; ++nt)
        #pragma unroll
        for (int v = 0; v < 8; ++v) {
          int m = mrow + v + 8*lh;
          dst[(size_t)(nt*16 + row0)*Sc + m] = (__bf16)acc[nt][v];
        }
    }
  }
}

// =====================================================================
// Kernel 2: causal flash attention per (b,h).
//   One wave owns a 16-query tile; loop over 32-key chunks, online softmax.
//   P transposed C->A layout via per-wave LDS buffer (same-wave in-order
//   LDS + s_wait_dscnt fence, no workgroup barrier -> causal divergence OK).
// =====================================================================
__global__ __launch_bounds__(128) void attn_kernel(
    const __bf16* __restrict__ Qb, const __bf16* __restrict__ Kb,
    const __bf16* __restrict__ Vt, __bf16* __restrict__ Ob)
{
  __shared__ __bf16 Plds[4][16*32];
  const int bh = blockIdx.y;
  const int b  = bh / Hc;
  const int h  = bh % Hc;
  const int tid = threadIdx.x, lane = tid & 31, wave = tid >> 5;
  const int row0 = lane & 15, lh = lane >> 4;
  const int qbase = blockIdx.x*64 + wave*16;
  __bf16* Pw = &Plds[wave][0];

  // Q A-tiles (bf16, two K=32 chunks)
  v16bf aq[2];
  const __bf16* qr = Qb + ((size_t)bh*Sc + qbase + row0) * HDc;
  #pragma unroll
  for (int kc = 0; kc < 2; ++kc) {
    const __bf16* p = qr + kc*32 + lh*8;
    #pragma unroll
    for (int e = 0; e < 8; ++e) { aq[kc][e] = p[e]; aq[kc][e+8] = p[16+e]; }
  }

  v8f acco[4] = {};
  float mrow[8], lrow[8];
  #pragma unroll
  for (int v = 0; v < 8; ++v) { mrow[v] = -1e30f; lrow[v] = 0.f; }
  const float scale = 0.125f;            // 1/sqrt(64)

  const int nchunks = (qbase + 16 + 31) >> 5;      // causal bound
  for (int jc = 0; jc < nchunks; ++jc) {
    const int jbase = jc * 32;

    // ---- scores: S[16x32] = Q[16x64] * K^T ----
    v8f accs[2] = {};
    #pragma unroll
    for (int nt = 0; nt < 2; ++nt) {
      const __bf16* kr = Kb + ((size_t)bh*Sc + jbase + nt*16 + row0) * HDc;
      #pragma unroll
      for (int kc = 0; kc < 2; ++kc) {
        const __bf16* p = kr + kc*32 + lh*8;
        v16bf bk;
        #pragma unroll
        for (int e = 0; e < 8; ++e) { bk[e] = p[e]; bk[e+8] = p[16+e]; }
        accs[nt] = wmma_bf16(aq[kc], bk, accs[nt]);
      }
    }

    // ---- scale + causal mask (C layout: row = v + 8*lh, col = row0) ----
    #pragma unroll
    for (int nt = 0; nt < 2; ++nt)
      #pragma unroll
      for (int v = 0; v < 8; ++v) {
        int q = qbase + v + 8*lh;
        int k = jbase + nt*16 + row0;
        float s = accs[nt][v] * scale;
        accs[nt][v] = (k <= q) ? s : -1e30f;
      }

    // ---- online softmax: per-row reduce across the 16-lane half ----
    float fscale[8];
    #pragma unroll
    for (int v = 0; v < 8; ++v) {
      float mx = fmaxf(accs[0][v], accs[1][v]);
      #pragma unroll
      for (int off = 1; off < 16; off <<= 1)
        mx = fmaxf(mx, __shfl_xor(mx, off, 32));
      float mnew = fmaxf(mrow[v], mx);
      float f    = __expf(mrow[v] - mnew);
      mrow[v]    = mnew;
      lrow[v]   *= f;
      fscale[v]  = f;
      float p0 = __expf(accs[0][v] - mnew);
      float p1 = __expf(accs[1][v] - mnew);
      accs[0][v] = p0; accs[1][v] = p1;
      float sum = p0 + p1;
      #pragma unroll
      for (int off = 1; off < 16; off <<= 1)
        sum += __shfl_xor(sum, off, 32);
      lrow[v] += sum;
    }
    #pragma unroll
    for (int t = 0; t < 4; ++t)
      #pragma unroll
      for (int v = 0; v < 8; ++v) acco[t][v] *= fscale[v];

    // ---- P: C layout -> LDS [16][32] bf16 -> A layout ----
    #pragma unroll
    for (int v = 0; v < 8; ++v) {
      int r = v + 8*lh;
      Pw[r*32 + row0]      = (__bf16)accs[0][v];
      Pw[r*32 + 16 + row0] = (__bf16)accs[1][v];
    }
    asm volatile("s_wait_dscnt 0" ::: "memory");   // same-wave LDS RAW fence
    v16bf ap;
    {
      const __bf16* p = Pw + row0*32 + lh*8;
      #pragma unroll
      for (int e = 0; e < 8; ++e) { ap[e] = p[e]; ap[e+8] = p[16+e]; }
    }
    asm volatile("" ::: "memory");

    // ---- O += P[16x32] * V[32x64]  (Vt rows are contiguous in s) ----
    #pragma unroll
    for (int t = 0; t < 4; ++t) {
      const __bf16* vr =
          Vt + (size_t)bh*HDc*Sc + (size_t)(t*16 + row0)*Sc + jbase + lh*8;
      v16bf bv;
      #pragma unroll
      for (int e = 0; e < 8; ++e) { bv[e] = vr[e]; bv[e+8] = vr[16+e]; }
      acco[t] = wmma_bf16(ap, bv, acco[t]);
    }
  }

  // ---- normalize, store head-concat bf16 [B,S,E] ----
  #pragma unroll
  for (int v = 0; v < 8; ++v) lrow[v] = 1.0f / lrow[v];
  #pragma unroll
  for (int t = 0; t < 4; ++t)
    #pragma unroll
    for (int v = 0; v < 8; ++v) {
      int s = qbase + v + 8*lh;
      Ob[((size_t)b*Sc + s)*Ec + h*HDc + t*16 + row0] =
          (__bf16)(acco[t][v] * lrow[v]);
    }
}

// =====================================================================
// Kernel 3: Wo -> Wo^T bf16 (B-operand layout for the projection GEMM)
// =====================================================================
__global__ __launch_bounds__(256) void wot_kernel(
    const float* __restrict__ Wo, __bf16* __restrict__ Wot)
{
  int i = blockIdx.x * 256 + threadIdx.x;          // over E*E
  int k = i >> 10, n = i & (Ec - 1);               // i = k*E + n
  Wot[(size_t)n*Ec + k] = (__bf16)Wo[i];
}

// =====================================================================
// Kernel 4: output projection GEMM  out[8192,1024] = Ob * Wo
//   Block 256 thr (8 waves): 128M x 64N tile; wave = 16M x 64N; K loop x32.
//   Wot is 2 MB -> L2-resident across all 1024 blocks.
// =====================================================================
__global__ __launch_bounds__(256) void proj_kernel(
    const __bf16* __restrict__ Ob,
    const __bf16* __restrict__ Wot,
    float* __restrict__ out)
{
  const int tid = threadIdx.x, lane = tid & 31, wave = tid >> 5;
  const int row0 = lane & 15, lh = lane >> 4;
  const int mbase = blockIdx.x*128 + wave*16;
  const int nbase = blockIdx.y*64;

  v8f acc[4] = {};
  const __bf16* ar = Ob + ((size_t)mbase + row0) * Ec;

  for (int kc = 0; kc < Ec/32; ++kc) {
    if (kc + 1 < Ec/32)                            // gfx1250 global_prefetch
      __builtin_prefetch(ar + (kc+1)*32, 0, 3);
    v16bf a;
    {
      const __bf16* p = ar + kc*32 + lh*8;
      #pragma unroll
      for (int e = 0; e < 8; ++e) { a[e] = p[e]; a[e+8] = p[16+e]; }
    }
    #pragma unroll
    for (int t = 0; t < 4; ++t) {
      const __bf16* q = Wot + (size_t)(nbase + t*16 + row0)*Ec + kc*32 + lh*8;
      v16bf bw;
      #pragma unroll
      for (int e = 0; e < 8; ++e) { bw[e] = q[e]; bw[e+8] = q[16+e]; }
      acc[t] = wmma_bf16(a, bw, acc[t]);
    }
  }
  #pragma unroll
  for (int t = 0; t < 4; ++t)
    #pragma unroll
    for (int v = 0; v < 8; ++v)
      out[(size_t)(mbase + v + 8*lh)*Ec + nbase + t*16 + row0] = acc[t][v];
}

// =====================================================================
extern "C" void kernel_launch(void* const* d_in, const int* in_sizes, int n_in,
                              void* d_out, int out_size, void* d_ws, size_t ws_size,
                              hipStream_t stream)
{
  const float* x  = (const float*)d_in[0];
  const float* Wq = (const float*)d_in[1];
  const float* Wk = (const float*)d_in[2];
  const float* Wv = (const float*)d_in[3];
  const float* Wo = (const float*)d_in[4];
  float* out = (float*)d_out;

  // workspace: Qb/Kb (16 MB ea), Vt (16 MB), Ob (16 MB), Wot (2 MB) = 66 MB
  __bf16* Qb  = (__bf16*)d_ws;
  __bf16* Kb  = Qb + (size_t)Bc*Hc*Sc*HDc;
  __bf16* Vt  = Kb + (size_t)Bc*Hc*Sc*HDc;
  __bf16* Ob  = Vt + (size_t)Bc*Hc*Sc*HDc;
  __bf16* Wot = Ob + (size_t)Bc*Sc*Ec;

  qkv_kernel<<<dim3(Sc/64, Bc*Hc), 128, 0, stream>>>(x, Wq, Wk, Wv, Qb, Kb, Vt);
  wot_kernel<<<dim3((Ec*Ec)/256), 256, 0, stream>>>(Wo, Wot);
  attn_kernel<<<dim3(Sc/64, Bc*Hc), 128, 0, stream>>>(Qb, Kb, Vt, Ob);
  proj_kernel<<<dim3((Bc*Sc)/128, Ec/64), 256, 0, stream>>>(Ob, Wot, out);
}
